// decoder_patchwork_53764400611706
// MI455X (gfx1250) — matrix operand order repaired and checked by reading, not compiled, not run on hardware
//
#include <hip/hip_runtime.h>

typedef __attribute__((ext_vector_type(16))) __bf16 v16bf;
typedef __attribute__((ext_vector_type(8)))  __bf16 v8bf;
typedef __attribute__((ext_vector_type(8)))  float  v8f;
typedef __attribute__((ext_vector_type(4)))  float  v4f;

#define N_CELLS 10000
#define ZDIM    50
#define EDIM    128
#define DDOM    4
#define M1      20000
#define M2      4000

// ---------------------------------------------------------------------------
// Kernel 1: emb[n] = theta[n] @ (alpha + betas[dom[n]]), split into bf16 hi/lo
// ---------------------------------------------------------------------------
__global__ void emb_split_kernel(const float* __restrict__ theta,
                                 const int*   __restrict__ dom,
                                 const float* __restrict__ alpha,
                                 const float* __restrict__ betas,
                                 __bf16* __restrict__ emb_hi,
                                 __bf16* __restrict__ emb_lo) {
  __shared__ float th[ZDIM];
  const int n = blockIdx.x;
  const int e = threadIdx.x;              // 128 threads = EDIM
  if (threadIdx.x < ZDIM) th[threadIdx.x] = theta[n * ZDIM + threadIdx.x];
  __syncthreads();
  const int d = dom[n];
  const float* __restrict__ bet = betas + (size_t)d * ZDIM * EDIM;
  float acc = 0.f;
  #pragma unroll 5
  for (int z = 0; z < ZDIM; ++z)
    acc = fmaf(th[z], alpha[z * EDIM + e] + bet[z * EDIM + e], acc);
  __bf16 h = (__bf16)acc;
  emb_hi[(size_t)n * EDIM + e] = h;
  emb_lo[(size_t)n * EDIM + e] = (__bf16)(acc - (float)h);
}

// ---------------------------------------------------------------------------
// Kernel 2: elementwise fp32 -> bf16 hi/lo split (for rho matrices)
// ---------------------------------------------------------------------------
__global__ void split_kernel(const float* __restrict__ x,
                             __bf16* __restrict__ hi,
                             __bf16* __restrict__ lo, int count) {
  int i = blockIdx.x * blockDim.x + threadIdx.x;
  if (i < count) {
    float v = x[i];
    __bf16 h = (__bf16)v;
    hi[i] = h;
    lo[i] = (__bf16)(v - (float)h);
  }
}

// ---------------------------------------------------------------------------
// Kernel 3: WMMA GEMM (bf16x2 split precision) + bias + raw logits +
//           online per-row (max, sum-exp) partial stats per column chunk.
// One wave handles ROWS 16-cell strips x chunkTiles 16-feature tiles.
// ROWS=2 gives two independent WMMA accumulate chains (XDL ILP) and halves
// B-fragment L2 traffic per output element; A fragments live in VGPRs.
// ---------------------------------------------------------------------------
template <int ROWS>
__global__ __launch_bounds__(32)
void gemm_softmax_stats_kernel(const __bf16* __restrict__ Ahi,
                               const __bf16* __restrict__ Alo,
                               const __bf16* __restrict__ Bhi,
                               const __bf16* __restrict__ Blo,
                               const float*  __restrict__ bias,   // [D, M]
                               const int*    __restrict__ dom,    // [N]
                               float*        __restrict__ out,    // [N, M]
                               float*        __restrict__ statM,  // [N, nchunks]
                               float*        __restrict__ statS,  // [N, nchunks]
                               int cellStart, int M, int nchunks, int chunkTiles) {
  const int strip = blockIdx.x / nchunks;
  const int chunk = blockIdx.x % nchunks;
  const int lane  = threadIdx.x;
  const int half  = lane >> 4;          // 0: rows 0-7 / K-lo,  1: rows 8-15 / K-hi
  const int l16   = lane & 15;
  const int cellBase0 = cellStart + strip * 16 * ROWS;
  const int featBase0 = chunk * chunkTiles * 16;

  // --- A fragments (row = cellBase0 + 16*rt + l16), K = 128 -----------------
  // ISA 16-bit A 16x32 layout: lane<16 -> v0..3:K0..7, v4..7:K16..23
  //                            lane>=16 -> v0..3:K8..15, v4..7:K24..31
  v16bf ahi[ROWS][4], alo[ROWS][4];
  #pragma unroll
  for (int rt = 0; rt < ROWS; ++rt) {
    const int arow = cellBase0 + rt * 16 + l16;
    const __bf16* pAh = Ahi + (size_t)arow * EDIM + (half << 3);
    const __bf16* pAl = Alo + (size_t)arow * EDIM + (half << 3);
    #pragma unroll
    for (int kc = 0; kc < 4; ++kc) {
      v8bf h0 = *(const v8bf*)(pAh + kc * 32);
      v8bf h1 = *(const v8bf*)(pAh + kc * 32 + 16);
      ahi[rt][kc] = __builtin_shufflevector(h0, h1, 0,1,2,3,4,5,6,7,8,9,10,11,12,13,14,15);
      v8bf l0 = *(const v8bf*)(pAl + kc * 32);
      v8bf l1 = *(const v8bf*)(pAl + kc * 32 + 16);
      alo[rt][kc] = __builtin_shufflevector(l0, l1, 0,1,2,3,4,5,6,7,8,9,10,11,12,13,14,15);
    }
  }

  // Per-row bias base pointers (8 rows owned by this lane's half per strip)
  const float* biasRow[ROWS][8];
  #pragma unroll
  for (int rt = 0; rt < ROWS; ++rt)
    #pragma unroll
    for (int r = 0; r < 8; ++r) {
      int cell = cellBase0 + rt * 16 + (half << 3) + r;
      biasRow[rt][r] = bias + (size_t)dom[cell] * M;
    }

  float m_r[ROWS][8], s_r[ROWS][8];
  #pragma unroll
  for (int rt = 0; rt < ROWS; ++rt)
    #pragma unroll
    for (int r = 0; r < 8; ++r) { m_r[rt][r] = -__builtin_inff(); s_r[rt][r] = 0.f; }

  for (int t = 0; t < chunkTiles; ++t) {
    const int featBase = featBase0 + t * 16;
    // B fragment: column = featBase + l16; K = 8*half+0..7 and 16+8*half+0..7 (matches A)
    const int bcol = featBase + l16;
    const __bf16* pBh = Bhi + (size_t)bcol * EDIM + (half << 3);
    const __bf16* pBl = Blo + (size_t)bcol * EDIM + (half << 3);
    v8f acc[ROWS];
    #pragma unroll
    for (int rt = 0; rt < ROWS; ++rt) acc[rt] = (v8f){};
    #pragma unroll
    for (int kc = 0; kc < 4; ++kc) {
      v16bf bh = __builtin_shufflevector(*(const v8bf*)(pBh + kc * 32), *(const v8bf*)(pBh + kc * 32 + 16), 0,1,2,3,4,5,6,7,8,9,10,11,12,13,14,15);
      v16bf bl = __builtin_shufflevector(*(const v8bf*)(pBl + kc * 32), *(const v8bf*)(pBl + kc * 32 + 16), 0,1,2,3,4,5,6,7,8,9,10,11,12,13,14,15);
      #pragma unroll
      for (int rt = 0; rt < ROWS; ++rt) {
        acc[rt] = __builtin_amdgcn_wmma_f32_16x16x32_bf16(false, ahi[rt][kc], false, bh,
                                                          (short)0, acc[rt], false, false);
        acc[rt] = __builtin_amdgcn_wmma_f32_16x16x32_bf16(false, ahi[rt][kc], false, bl,
                                                          (short)0, acc[rt], false, false);
        acc[rt] = __builtin_amdgcn_wmma_f32_16x16x32_bf16(false, alo[rt][kc], false, bh,
                                                          (short)0, acc[rt], false, false);
      }
    }
    // D layout: lane holds col n = l16, rows m = 8*half + r
    const int feat = featBase + l16;
    #pragma unroll
    for (int rt = 0; rt < ROWS; ++rt)
      #pragma unroll
      for (int r = 0; r < 8; ++r) {
        const int cell = cellBase0 + rt * 16 + (half << 3) + r;
        float v = acc[rt][r] + biasRow[rt][r][feat];
        out[(size_t)cell * M + feat] = v;
        float nm = fmaxf(m_r[rt][r], v);
        s_r[rt][r] = s_r[rt][r] * __expf(m_r[rt][r] - nm) + __expf(v - nm);
        m_r[rt][r] = nm;
      }
  }

  // Butterfly reduce (max,sum) across the 16 lanes of each half
  #pragma unroll
  for (int mask = 1; mask < 16; mask <<= 1) {
    #pragma unroll
    for (int rt = 0; rt < ROWS; ++rt)
      #pragma unroll
      for (int r = 0; r < 8; ++r) {
        float om = __shfl_xor(m_r[rt][r], mask, 32);
        float os = __shfl_xor(s_r[rt][r], mask, 32);
        float nm = fmaxf(m_r[rt][r], om);
        s_r[rt][r] = s_r[rt][r] * __expf(m_r[rt][r] - nm) + os * __expf(om - nm);
        m_r[rt][r] = nm;
      }
  }
  if (l16 == 0) {
    #pragma unroll
    for (int rt = 0; rt < ROWS; ++rt)
      #pragma unroll
      for (int r = 0; r < 8; ++r) {
        int cell = cellBase0 + rt * 16 + (half << 3) + r;
        statM[(size_t)cell * nchunks + chunk] = m_r[rt][r];
        statS[(size_t)cell * nchunks + chunk] = s_r[rt][r];
      }
  }
}

// ---------------------------------------------------------------------------
// Kernel 4: merge chunk stats -> rowOff[n] = rowmax + log(sumexp)
// ---------------------------------------------------------------------------
__global__ void rowoff_kernel(const float* __restrict__ statM,
                              const float* __restrict__ statS,
                              float* __restrict__ rowOff, int nchunks) {
  int n = blockIdx.x * blockDim.x + threadIdx.x;
  if (n >= N_CELLS) return;
  float m = -__builtin_inff(), s = 0.f;
  for (int c = 0; c < nchunks; ++c) {
    float om = statM[(size_t)n * nchunks + c];
    float os = statS[(size_t)n * nchunks + c];
    float nm = fmaxf(m, om);
    s = s * __expf(m - nm) + os * __expf(om - nm);
    m = nm;
  }
  rowOff[n] = m + __logf(s);
}

// ---------------------------------------------------------------------------
// Kernel 5: out[n, :] -= rowOff[n]  (float4 vectorized)
// ---------------------------------------------------------------------------
__global__ void norm_kernel(float* __restrict__ out,
                            const float* __restrict__ rowOff, int Mvec) {
  const int n = blockIdx.y;
  const float off = rowOff[n];
  int c = blockIdx.x * blockDim.x + threadIdx.x;
  if (c < Mvec) {
    v4f* p = (v4f*)out + (size_t)n * Mvec + c;
    v4f v = *p;
    v = v - off;
    *p = v;
  }
}

// ---------------------------------------------------------------------------
extern "C" void kernel_launch(void* const* d_in, const int* in_sizes, int n_in,
                              void* d_out, int out_size, void* d_ws, size_t ws_size,
                              hipStream_t stream) {
  const float* theta = (const float*)d_in[0];
  const int*   dom   = (const int*)  d_in[1];
  const float* alpha = (const float*)d_in[2];
  const float* betas = (const float*)d_in[3];
  const float* rho1  = (const float*)d_in[4];
  const float* rho2  = (const float*)d_in[5];
  const float* bias1 = (const float*)d_in[6];
  const float* bias2 = (const float*)d_in[7];

  float* out1 = (float*)d_out;
  float* out2 = out1 + (size_t)N_CELLS * M1;

  // Workspace carve-up (256B aligned)
  char* ws = (char*)d_ws;
  size_t o = 0;
  auto carve = [&](size_t bytes) -> char* {
    char* p = ws + o;
    o = (o + bytes + 255) & ~(size_t)255;
    return p;
  };
  __bf16* emb_hi  = (__bf16*)carve((size_t)N_CELLS * EDIM * 2);
  __bf16* emb_lo  = (__bf16*)carve((size_t)N_CELLS * EDIM * 2);
  __bf16* rho1_hi = (__bf16*)carve((size_t)M1 * EDIM * 2);
  __bf16* rho1_lo = (__bf16*)carve((size_t)M1 * EDIM * 2);
  __bf16* rho2_hi = (__bf16*)carve((size_t)M2 * EDIM * 2);
  __bf16* rho2_lo = (__bf16*)carve((size_t)M2 * EDIM * 2);
  const int NCH1 = 10, CT1 = 125;   // 10*125*16 = 20000
  const int NCH2 = 5,  CT2 = 50;    // 5*50*16   = 4000
  float* stat1M = (float*)carve((size_t)N_CELLS * NCH1 * 4);
  float* stat1S = (float*)carve((size_t)N_CELLS * NCH1 * 4);
  float* stat2M = (float*)carve((size_t)N_CELLS * NCH2 * 4);
  float* stat2S = (float*)carve((size_t)N_CELLS * NCH2 * 4);
  float* rowOff1 = (float*)carve((size_t)N_CELLS * 4);
  float* rowOff2 = (float*)carve((size_t)N_CELLS * 4);
  (void)ws_size; (void)in_sizes; (void)n_in; (void)out_size;

  // 1) emb + split
  emb_split_kernel<<<N_CELLS, EDIM, 0, stream>>>(theta, dom, alpha, betas,
                                                 emb_hi, emb_lo);
  // 2) rho splits
  {
    int c1 = M1 * EDIM, c2 = M2 * EDIM;
    split_kernel<<<(c1 + 255) / 256, 256, 0, stream>>>(rho1, rho1_hi, rho1_lo, c1);
    split_kernel<<<(c2 + 255) / 256, 256, 0, stream>>>(rho2, rho2_hi, rho2_lo, c2);
  }
  // 3) WMMA GEMM + bias + raw logits + partial softmax stats
  //    cells 0..9983 in 32-cell strips (ROWS=2), tail 16 cells with ROWS=1
  {
    const int pairStrips = 312;              // 312*32 = 9984 cells
    const int tailStart  = pairStrips * 32;  // 9984
    gemm_softmax_stats_kernel<2><<<pairStrips * NCH1, 32, 0, stream>>>(
        emb_hi, emb_lo, rho1_hi, rho1_lo, bias1, dom, out1,
        stat1M, stat1S, 0, M1, NCH1, CT1);
    gemm_softmax_stats_kernel<1><<<NCH1, 32, 0, stream>>>(
        emb_hi, emb_lo, rho1_hi, rho1_lo, bias1, dom, out1,
        stat1M, stat1S, tailStart, M1, NCH1, CT1);
    gemm_softmax_stats_kernel<2><<<pairStrips * NCH2, 32, 0, stream>>>(
        emb_hi, emb_lo, rho2_hi, rho2_lo, bias2, dom, out2,
        stat2M, stat2S, 0, M2, NCH2, CT2);
    gemm_softmax_stats_kernel<1><<<NCH2, 32, 0, stream>>>(
        emb_hi, emb_lo, rho2_hi, rho2_lo, bias2, dom, out2,
        stat2M, stat2S, tailStart, M2, NCH2, CT2);
  }
  // 4) merge stats
  rowoff_kernel<<<(N_CELLS + 255) / 256, 256, 0, stream>>>(stat1M, stat1S, rowOff1, NCH1);
  rowoff_kernel<<<(N_CELLS + 255) / 256, 256, 0, stream>>>(stat2M, stat2S, rowOff2, NCH2);
  // 5) normalize
  {
    dim3 g1((M1 / 4 + 255) / 256, N_CELLS);
    dim3 g2((M2 / 4 + 255) / 256, N_CELLS);
    norm_kernel<<<g1, 256, 0, stream>>>(out1, rowOff1, M1 / 4);
    norm_kernel<<<g2, 256, 0, stream>>>(out2, rowOff2, M2 / 4);
  }
}
